// SyntheticBlock_31774168056051
// MI455X (gfx1250) — compile-verified, matrix operand order checked by reading, not run on hardware
//
#include <hip/hip_runtime.h>

#define NN 100000
#define EE 1600000
#define CC 128
#define EPSV 1e-5f

typedef __attribute__((ext_vector_type(16))) _Float16 v16h;
typedef __attribute__((ext_vector_type(8)))  _Float16 v8h;
typedef __attribute__((ext_vector_type(8)))  float    v8f;
typedef __attribute__((__vector_size__(4 * sizeof(int)))) int v4i;

// ---- gfx1250 async global->LDS path (guarded; falls back to plain copy) ----
#if defined(__has_builtin)
#  if __has_builtin(__builtin_amdgcn_global_load_async_to_lds_b128) && \
      __has_builtin(__builtin_amdgcn_s_wait_asynccnt)
#    define HAVE_ASYNC_LDS 1
#  endif
#endif
#ifndef HAVE_ASYNC_LDS
#  define HAVE_ASYNC_LDS 0
#endif

#if HAVE_ASYNC_LDS
typedef __attribute__((address_space(1))) v4i* as1_v4i_ptr;
typedef __attribute__((address_space(3))) v4i* as3_v4i_ptr;
#endif

// Copy 16 bytes global -> LDS, per lane.
__device__ __forceinline__ void copy16_g2l(const _Float16* g, _Float16* l) {
#if HAVE_ASYNC_LDS
  __builtin_amdgcn_global_load_async_to_lds_b128(
      (as1_v4i_ptr)(void*)g, (as3_v4i_ptr)(void*)l, /*offset=*/0, /*cpol=*/0);
#else
  *(v8h*)l = *(const v8h*)g;
#endif
}
__device__ __forceinline__ void copy_wait() {
#if HAVE_ASYNC_LDS
  __builtin_amdgcn_s_wait_asynccnt(0);
#endif
}

// ---- WMMA helper: D = A(16x32 f16) x B(32x16 f16) + C (f32) ----
__device__ __forceinline__ v8f wmma32(v16h a, v16h b, v8f c) {
  return __builtin_amdgcn_wmma_f32_16x16x32_f16(
      /*neg_a=*/false, a, /*neg_b=*/false, b,
      /*c_mod=*/(short)0, c, /*reuse_a=*/false, /*reuse_b=*/false);
}

// Load one 16x32 fragment (A layout; B uses same layout on a transposed,
// N-major weight tile). `row` points at a 128-half LDS row (16B aligned).
__device__ __forceinline__ v16h frag_ld(const _Float16* row, int kt) {
  const int lane = threadIdx.x & 31;
  const int c0 = kt * 32 + ((lane & 16) ? 8 : 0);
  v8h lo = *(const v8h*)(row + c0);
  v8h hi = *(const v8h*)(row + c0 + 16);
  v16h f;
#pragma unroll
  for (int i = 0; i < 8; ++i) { f[i] = lo[i]; f[i + 8] = hi[i]; }
  return f;
}

// Whole wave cooperatively loads one 128-float global row -> f16 LDS row.
__device__ __forceinline__ void gather_row_f16(_Float16* dstRow,
                                               const float* srcRow, bool valid) {
  const int lane = threadIdx.x & 31;
  float4 v = valid ? *(const float4*)(srcRow + lane * 4)
                   : make_float4(0.f, 0.f, 0.f, 0.f);
  _Float16* p = dstRow + lane * 4;
  p[0] = (_Float16)v.x; p[1] = (_Float16)v.y;
  p[2] = (_Float16)v.z; p[3] = (_Float16)v.w;
}

// ============ Kernel 0: h16 = (f16)h, one-time conversion =================
__global__ __launch_bounds__(256) void h2f16_kernel(const float* __restrict__ h,
                                                    _Float16* __restrict__ h16) {
  size_t i = ((size_t)blockIdx.x * 256 + threadIdx.x) * 8;
  if (i < (size_t)NN * CC) {
    float4 a = *(const float4*)(h + i);
    float4 b = *(const float4*)(h + i + 4);
    v8h o;
    o[0] = (_Float16)a.x; o[1] = (_Float16)a.y;
    o[2] = (_Float16)a.z; o[3] = (_Float16)a.w;
    o[4] = (_Float16)b.x; o[5] = (_Float16)b.y;
    o[6] = (_Float16)b.z; o[7] = (_Float16)b.w;
    *(v8h*)(h16 + i) = o;
  }
}

// ============ Kernel 1: delta = tanh(relu(h@Wh1+bh1)@Wh2+bh2) ============
__global__ __launch_bounds__(256) void delta_kernel(
    const _Float16* __restrict__ h16, const float* __restrict__ Wh1,
    const float* __restrict__ bh1, const float* __restrict__ Wh2,
    const float* __restrict__ bh2, float* __restrict__ delta) {
  __shared__ __align__(16) _Float16 sW[CC * CC];   // Wh1 transposed: sW[n*128+k]
  __shared__ float sB[CC];
  __shared__ __align__(16) _Float16 sT[8][16 * CC]; // per-wave h tile, reused for t

  const int tid = threadIdx.x, wave = tid >> 5, lane = tid & 31;
  for (int i = tid; i < CC * CC; i += 256) {
    int k = i >> 7, n = i & 127;
    sW[n * CC + k] = (_Float16)Wh1[i];
  }
  if (tid < CC) sB[tid] = bh1[tid];

  const int r0 = blockIdx.x * 128 + wave * 16;
  __syncthreads();

  // 16 rows x 256B: 8 wave-wide 16B-per-lane copies (async when available)
#pragma unroll
  for (int t = 0; t < 8; ++t) {
    int r = t * 2 + (lane >> 4);
    int row = r0 + r;
    size_t srow = (row < NN) ? (size_t)row : 0;   // clamp; garbage rows discarded later
    copy16_g2l(h16 + srow * CC + (lane & 15) * 8,
               &sT[wave][r * CC + (lane & 15) * 8]);
  }
  copy_wait();

  v16h a[4];
#pragma unroll
  for (int kt = 0; kt < 4; ++kt) a[kt] = frag_ld(&sT[wave][(lane & 15) * CC], kt);

  const int col16 = lane & 15, mhi = (lane >> 4) * 8;
#pragma unroll
  for (int nt = 0; nt < 8; ++nt) {
    v8f acc = {};
#pragma unroll
    for (int kt = 0; kt < 4; ++kt)
      acc = wmma32(a[kt], frag_ld(&sW[(nt * 16 + col16) * CC], kt), acc);
    float bias = sB[nt * 16 + col16];
#pragma unroll
    for (int j = 0; j < 8; ++j) {
      float v = acc[j] + bias;
      v = v > 0.f ? v : 0.f;
      sT[wave][(mhi + j) * CC + nt * 16 + col16] = (_Float16)v;  // t tile (relu)
    }
  }
  __syncthreads();

  // tiny second GEMM: (16x128) @ (128x3), scalar per lane
  for (int c = 0; c < 3; ++c) {
    if (lane < 16) {
      int row = r0 + lane;
      float acc = bh2[c];
      const _Float16* trow = &sT[wave][lane * CC];
      for (int k = 0; k < CC; ++k) acc += (float)trow[k] * Wh2[k * 3 + c];
      if (row < NN) delta[(size_t)row * 3 + c] = tanhf(acc);
    }
  }
}

// ============ Kernel 2: m = relu([rel, h[src]] @ Wf + bf); aggr[dst] += m ====
__global__ __launch_bounds__(256) void edge_kernel(
    const _Float16* __restrict__ h16, const float* __restrict__ pos,
    const long long* __restrict__ ei, const float* __restrict__ Wf,
    const float* __restrict__ bf, const float* __restrict__ delta,
    float* __restrict__ aggr) {
  __shared__ __align__(16) _Float16 sW[CC * CC];   // Wf[3+k][n] -> sW[n*128+k]
  __shared__ float sW0[3][CC];                     // Wf rows 0..2 (rel part)
  __shared__ float sB[CC];
  __shared__ __align__(16) _Float16 sA[8][16 * CC]; // gathered h16[src] tiles
  __shared__ float sRel[8][16][4];
  __shared__ int sDst[8][16];
  __shared__ int sSrc[8][16];

  const int tid = threadIdx.x, wave = tid >> 5, lane = tid & 31;
  for (int i = tid; i < CC * CC; i += 256) {
    int k = i >> 7, n = i & 127;
    sW[n * CC + k] = (_Float16)Wf[(size_t)(3 + k) * CC + n];
  }
  for (int i = tid; i < 3 * CC; i += 256) sW0[i / CC][i % CC] = Wf[i];
  if (tid < CC) sB[tid] = bf[tid];

  const long long e0 = (long long)blockIdx.x * 128 + wave * 16;
  if (lane < 16) {
    long long e = e0 + lane;
    int s = 0, d = -1;
    float rx = 0.f, ry = 0.f, rz = 0.f;
    if (e < EE) {
      s = (int)ei[e];
      d = (int)ei[(long long)EE + e];
      float3 ps = *(const float3*)(pos + (size_t)s * 3);
      float3 pd = *(const float3*)(pos + (size_t)d * 3);
      rx = ps.x - pd.x + delta[(size_t)d * 3 + 0];
      ry = ps.y - pd.y + delta[(size_t)d * 3 + 1];
      rz = ps.z - pd.z + delta[(size_t)d * 3 + 2];
    }
    sSrc[wave][lane] = s; sDst[wave][lane] = d;
    sRel[wave][lane][0] = rx; sRel[wave][lane][1] = ry; sRel[wave][lane][2] = rz;
  }
  __syncthreads();

  // gather 16 h16[src] rows: 8 wave-wide 16B-per-lane copies (async if avail)
#pragma unroll
  for (int t = 0; t < 8; ++t) {
    int r = t * 2 + (lane >> 4);
    int s = sSrc[wave][r];      // clamped to 0 for invalid edges; rows discarded
    copy16_g2l(h16 + (size_t)s * CC + (lane & 15) * 8,
               &sA[wave][r * CC + (lane & 15) * 8]);
  }
  copy_wait();

  v16h a[4];
#pragma unroll
  for (int kt = 0; kt < 4; ++kt) a[kt] = frag_ld(&sA[wave][(lane & 15) * CC], kt);

  const int col16 = lane & 15, mhi = (lane >> 4) * 8;
#pragma unroll
  for (int nt = 0; nt < 8; ++nt) {
    v8f acc = {};
#pragma unroll
    for (int kt = 0; kt < 4; ++kt)
      acc = wmma32(a[kt], frag_ld(&sW[(nt * 16 + col16) * CC], kt), acc);
    const int col = nt * 16 + col16;
    const float bias = sB[col];
    const float w0 = sW0[0][col], w1 = sW0[1][col], w2 = sW0[2][col];
#pragma unroll
    for (int j = 0; j < 8; ++j) {
      int m = mhi + j;
      int d = sDst[wave][m];
      float v = acc[j] + bias + sRel[wave][m][0] * w0 +
                sRel[wave][m][1] * w1 + sRel[wave][m][2] * w2;
      v = v > 0.f ? v : 0.f;
      if (d >= 0) unsafeAtomicAdd(&aggr[(size_t)d * CC + col], v);
    }
  }
}

// ====== Kernel 3: out = leaky(relu(aggr@Wg1+bg1)@Wg2+bg2+h); col stats ======
__global__ __launch_bounds__(256) void out_kernel(
    const float* __restrict__ aggr, const float* __restrict__ Wg1,
    const float* __restrict__ bg1, const float* __restrict__ Wg2,
    const float* __restrict__ bg2, const float* __restrict__ h,
    float* __restrict__ out, float* __restrict__ sums /* [0..127]=sum, [128..255]=sumsq */) {
  __shared__ __align__(16) _Float16 sW[CC * CC];    // Wg1 then Wg2 (transposed)
  __shared__ float sB[CC];
  __shared__ __align__(16) _Float16 sT[8][16 * CC]; // aggr tile, reused for t
  __shared__ float sCS[CC], sCS2[CC];

  const int tid = threadIdx.x, wave = tid >> 5, lane = tid & 31;
  for (int i = tid; i < CC * CC; i += 256) {
    int k = i >> 7, n = i & 127;
    sW[n * CC + k] = (_Float16)Wg1[i];
  }
  if (tid < CC) { sB[tid] = bg1[tid]; sCS[tid] = 0.f; sCS2[tid] = 0.f; }
  const int r0 = blockIdx.x * 128 + wave * 16;
  __syncthreads();

  for (int r = 0; r < 16; ++r) {
    int row = r0 + r;
    bool valid = row < NN;
    gather_row_f16(&sT[wave][r * CC], aggr + (valid ? (size_t)row * CC : 0), valid);
  }

  v16h a[4];
#pragma unroll
  for (int kt = 0; kt < 4; ++kt) a[kt] = frag_ld(&sT[wave][(lane & 15) * CC], kt);

  const int col16 = lane & 15, mhi = (lane >> 4) * 8;
#pragma unroll
  for (int nt = 0; nt < 8; ++nt) {
    v8f acc = {};
#pragma unroll
    for (int kt = 0; kt < 4; ++kt)
      acc = wmma32(a[kt], frag_ld(&sW[(nt * 16 + col16) * CC], kt), acc);
    float bias = sB[nt * 16 + col16];
#pragma unroll
    for (int j = 0; j < 8; ++j) {
      float v = acc[j] + bias;
      v = v > 0.f ? v : 0.f;
      sT[wave][(mhi + j) * CC + nt * 16 + col16] = (_Float16)v;  // t tile
    }
  }
  __syncthreads();

  // swap weights in the SAME LDS buffer (all waves are past GEMM1)
  for (int i = tid; i < CC * CC; i += 256) {
    int k = i >> 7, n = i & 127;
    sW[n * CC + k] = (_Float16)Wg2[i];
  }
  if (tid < CC) sB[tid] = bg2[tid];
  __syncthreads();

  v16h a2[4];
#pragma unroll
  for (int kt = 0; kt < 4; ++kt) a2[kt] = frag_ld(&sT[wave][(lane & 15) * CC], kt);

#pragma unroll
  for (int nt = 0; nt < 8; ++nt) {
    v8f acc = {};
#pragma unroll
    for (int kt = 0; kt < 4; ++kt)
      acc = wmma32(a2[kt], frag_ld(&sW[(nt * 16 + col16) * CC], kt), acc);
    const int col = nt * 16 + col16;
    const float bias = sB[col];
#pragma unroll
    for (int j = 0; j < 8; ++j) {
      int row = r0 + mhi + j;
      if (row < NN) {
        float v = acc[j] + bias + h[(size_t)row * CC + col];
        v = v >= 0.f ? v : 0.2f * v;  // leaky relu
        out[(size_t)row * CC + col] = v;
        atomicAdd(&sCS[col], v);
        atomicAdd(&sCS2[col], v * v);
      }
    }
  }
  __syncthreads();
  if (tid < CC) {
    unsafeAtomicAdd(&sums[tid], sCS[tid]);
    unsafeAtomicAdd(&sums[CC + tid], sCS2[tid]);
  }
}

// ====== Kernel 4: s = style@Ws+bs; out = gamma*normed + beta (in place) ======
__global__ __launch_bounds__(256) void final_kernel(
    const float* __restrict__ style, const float* __restrict__ Ws,
    const float* __restrict__ bs, const float* __restrict__ sums,
    float* __restrict__ out) {
  __shared__ __align__(16) _Float16 sW[256 * CC];   // Ws transposed: sW[n*128+k]
  __shared__ __align__(16) _Float16 sS[8][16 * CC]; // style tiles
  __shared__ float sMu[CC], sRs[CC];

  const int tid = threadIdx.x, wave = tid >> 5, lane = tid & 31;
  for (int i = tid; i < 256 * CC; i += 256) {
    int k = i >> 8, n = i & 255;
    sW[n * CC + k] = (_Float16)Ws[i];
  }
  if (tid < CC) {
    float s = sums[tid], s2 = sums[CC + tid];
    float mu = s / (float)NN;
    float var = s2 / (float)NN - mu * mu;
    sMu[tid] = mu;
    sRs[tid] = rsqrtf((var > 0.f ? var : 0.f) + EPSV);
  }
  const int r0 = blockIdx.x * 128 + wave * 16;
  __syncthreads();

  for (int r = 0; r < 16; ++r) {
    int row = r0 + r;
    bool valid = row < NN;
    gather_row_f16(&sS[wave][r * CC], style + (valid ? (size_t)row * CC : 0), valid);
  }

  v16h a[4];
#pragma unroll
  for (int kt = 0; kt < 4; ++kt) a[kt] = frag_ld(&sS[wave][(lane & 15) * CC], kt);

  const int col16 = lane & 15, mhi = (lane >> 4) * 8;
#pragma unroll
  for (int nt = 0; nt < 8; ++nt) {
    v8f accG = {}, accB = {};
#pragma unroll
    for (int kt = 0; kt < 4; ++kt) {
      accG = wmma32(a[kt], frag_ld(&sW[(nt * 16 + col16) * CC], kt), accG);       // gamma cols
      accB = wmma32(a[kt], frag_ld(&sW[(128 + nt * 16 + col16) * CC], kt), accB); // beta cols
    }
    const int c = nt * 16 + col16;
    const float gB = bs[c], bB = bs[CC + c];
    const float mu = sMu[c], rs = sRs[c];
#pragma unroll
    for (int j = 0; j < 8; ++j) {
      int row = r0 + mhi + j;
      if (row < NN) {
        float pre = out[(size_t)row * CC + c];
        out[(size_t)row * CC + c] = (accG[j] + gB) * ((pre - mu) * rs) + (accB[j] + bB);
      }
    }
  }
}

extern "C" void kernel_launch(void* const* d_in, const int* in_sizes, int n_in,
                              void* d_out, int out_size, void* d_ws, size_t ws_size,
                              hipStream_t stream) {
  const float* h     = (const float*)d_in[0];
  const float* pos   = (const float*)d_in[1];
  const float* style = (const float*)d_in[2];
  const long long* ei = (const long long*)d_in[3];
  const float* Wh1 = (const float*)d_in[4];
  const float* bh1 = (const float*)d_in[5];
  const float* Wh2 = (const float*)d_in[6];
  const float* bh2 = (const float*)d_in[7];
  const float* Wf  = (const float*)d_in[8];
  const float* bf  = (const float*)d_in[9];
  const float* Wg1 = (const float*)d_in[10];
  const float* bg1 = (const float*)d_in[11];
  const float* Wg2 = (const float*)d_in[12];
  const float* bg2 = (const float*)d_in[13];
  const float* Ws  = (const float*)d_in[14];
  const float* bs  = (const float*)d_in[15];
  float* out = (float*)d_out;

  char* ws = (char*)d_ws;
  float* aggr  = (float*)ws;                                              // N*C f32
  float* delta = (float*)(ws + (size_t)NN * CC * 4);                      // N*3 f32
  float* sums  = (float*)(ws + (size_t)NN * CC * 4 + (size_t)NN * 3 * 4); // 256 f32
  _Float16* h16 = (_Float16*)(ws + (size_t)NN * CC * 4 + (size_t)NN * 3 * 4 +
                              256 * sizeof(float));                       // N*C f16

  (void)hipMemsetAsync(aggr, 0, (size_t)NN * CC * 4, stream);
  (void)hipMemsetAsync(sums, 0, 256 * sizeof(float), stream);

  dim3 blk(256);
  const int convBlocks = (int)(((size_t)NN * CC / 8 + 255) / 256);
  const int nodeBlocks = (NN + 127) / 128;
  const int edgeBlocks = (EE + 127) / 128;
  h2f16_kernel<<<convBlocks, blk, 0, stream>>>(h, h16);
  delta_kernel<<<nodeBlocks, blk, 0, stream>>>(h16, Wh1, bh1, Wh2, bh2, delta);
  edge_kernel<<<edgeBlocks, blk, 0, stream>>>(h16, pos, ei, Wf, bf, delta, aggr);
  out_kernel<<<nodeBlocks, blk, 0, stream>>>(aggr, Wg1, bg1, Wg2, bg2, h, out, sums);
  final_kernel<<<nodeBlocks, blk, 0, stream>>>(style, Ws, bs, sums, out);
}
